// BahdanauAttention_32057635898061
// MI455X (gfx1250) — matrix operand order repaired and checked by reading, not compiled, not run on hardware
//
#include <hip/hip_runtime.h>
#include <hip/hip_bf16.h>
#include <math.h>

// ---------------------------------------------------------------------------
// Bahdanau attention, B=32, S=4096, H=256, fused bf16-WMMA score kernel.
// ---------------------------------------------------------------------------

typedef __attribute__((ext_vector_type(16))) __bf16 v16bf;
typedef __attribute__((ext_vector_type(8)))  float  v8f;
typedef __attribute__((ext_vector_type(8)))  unsigned int v8u;

#define BA_B 32
#define BA_S 4096
#define BA_H 256

// Native CDNA5 V_TANH_F32 (single trans op, co-executes with the matrix pipe)
// instead of the branchy libm expansion.
#if __has_builtin(__builtin_amdgcn_tanhf)
__device__ __forceinline__ float fast_tanh(float x) { return __builtin_amdgcn_tanhf(x); }
#elif __has_builtin(__builtin_amdgcn_tanh_f32)
__device__ __forceinline__ float fast_tanh(float x) { return __builtin_amdgcn_tanh_f32(x); }
#else
__device__ __forceinline__ float fast_tanh(float x) { return tanhf(x); }
#endif

__device__ __forceinline__ unsigned short f2bf(float x) {
    union { float f; unsigned u; } v; v.f = x;
    unsigned r = v.u + 0x7FFFu + ((v.u >> 16) & 1u);   // round-to-nearest-even
    return (unsigned short)(r >> 16);
}

// ---------------------------------------------------------------------------
// K0: pack W2 (H x H fp32, row-major) into bf16 B-fragment layout.
// Entry (c, n, h) = 16 contiguous bf16: W2[c*32 + h*8 + (kk&7) + (kk>>3)*16, n], kk=0..15.
// Flat index f = c*8192 + n*32 + h*16 + kk.
// ---------------------------------------------------------------------------
__global__ void k_pack_w2(const float* __restrict__ W2, unsigned short* __restrict__ pw2) {
    int f  = blockIdx.x * 256 + threadIdx.x;          // 65536 entries
    int kk = f & 15;
    int h  = (f >> 4) & 1;
    int n  = (f >> 5) & 255;
    int c  = f >> 13;
    int krow = c * 32 + h * 8 + (kk & 7) + (kk >> 3) * 16;
    pw2[f] = f2bf(W2[krow * BA_H + n]);
}

// ---------------------------------------------------------------------------
// K1: qb2[b,n] = sum_h hidden[b,h] * W1_w[h,n] + W1_b[n] + W2_b[n]
// ---------------------------------------------------------------------------
__global__ void k_qb2(const float* __restrict__ hidden, const float* __restrict__ W1_w,
                      const float* __restrict__ W1_b, const float* __restrict__ W2_b,
                      float* __restrict__ qb2) {
    __shared__ float hrow[BA_H];
    int b = blockIdx.x, n = threadIdx.x;
    hrow[n] = hidden[b * BA_H + n];
    __syncthreads();
    float s = W1_b[n] + W2_b[n];
#pragma unroll 8
    for (int h = 0; h < BA_H; ++h)
        s += hrow[h] * W1_w[h * BA_H + n];
    qb2[b * BA_H + n] = s;
}

// ---------------------------------------------------------------------------
// K2: fused  scores[b,s] = sum_n tanh(qb2[b,n] + (enc@W2)[s,n]) * Vw[n] + Vb
// Grid: 1024 blocks x 256 threads (8 waves). Block -> 128 rows of one batch.
// Wave -> 16 rows. A (enc rows, fp32->bf16) kept in regs for all K=256.
// ---------------------------------------------------------------------------
__global__ void k_scores(const float* __restrict__ enc,
                         const unsigned short* __restrict__ pw2,
                         const float* __restrict__ qb2,
                         const float* __restrict__ Vw,
                         const float* __restrict__ Vb,
                         float* __restrict__ scores) {
    const int tid   = threadIdx.x;
    const int lane  = tid & 31;
    const int wv    = tid >> 5;                 // wave 0..7
    const int lan16 = lane & 15;
    const int half  = lane >> 4;                // 0: lanes 0-15, 1: lanes 16-31
    const int b     = blockIdx.x >> 5;          // 32 blocks per batch
    const int sbase = (blockIdx.x & 31) * 128;  // row base within batch

    // ---- load & pack this wave's 16-row A tile for all K=256 (bf16) ----
    const float* rowp = enc + ((size_t)b * BA_S + sbase + wv * 16 + lan16) * BA_H;
    v8u apk[8];
#pragma unroll
    for (int c = 0; c < 8; ++c) {
        int o0 = c * 32 + half * 8;             // 16-bit A layout (ISA 7.12.2)
        float4 x0 = *(const float4*)(rowp + o0);
        float4 x1 = *(const float4*)(rowp + o0 + 4);
        float4 x2 = *(const float4*)(rowp + o0 + 16);
        float4 x3 = *(const float4*)(rowp + o0 + 20);
        float f[16] = { x0.x, x0.y, x0.z, x0.w, x1.x, x1.y, x1.z, x1.w,
                        x2.x, x2.y, x2.z, x2.w, x3.x, x3.y, x3.z, x3.w };
#pragma unroll
        for (int j = 0; j < 8; ++j)
            apk[c][j] = (unsigned)f2bf(f[2 * j]) | ((unsigned)f2bf(f[2 * j + 1]) << 16);
    }

    const float Vb0 = Vb[0];
    float spart[8] = {0.f, 0.f, 0.f, 0.f, 0.f, 0.f, 0.f, 0.f};

    const char* pw2b = (const char*)pw2;
    for (int t = 0; t < 16; ++t) {              // N tiles of 16
        const int n = t * 16 + lan16;
        const float qv = qb2[b * BA_H + n];
        const float vw = Vw[n];

        v8f acc = {0.f, 0.f, 0.f, 0.f, 0.f, 0.f, 0.f, 0.f};
#pragma unroll
        for (int c = 0; c < 8; ++c) {           // K chunks of 32
            // B fragment: 32 contiguous bytes, one 32B load per lane
            size_t boff = ((size_t)((c * 256 + n) * 2 + half)) * 32;
            v8u bu = *(const v8u*)(pw2b + boff);
            acc = __builtin_amdgcn_wmma_f32_16x16x32_bf16(
                false, __builtin_bit_cast(v16bf, apk[c]),
                false, __builtin_bit_cast(v16bf, bu),
                (short)0, acc, false, false);
        }
        // C/D layout: vgpr i, lanes0-15 -> M=i, lanes16-31 -> M=8+i; N = lane%16
#pragma unroll
        for (int i = 0; i < 8; ++i)
            spart[i] += fast_tanh(acc[i] + qv) * vw;
    }

    // reduce across the 16 lanes sharing the same rows (masks stay in half)
#pragma unroll
    for (int i = 0; i < 8; ++i) {
        float v = spart[i];
        v += __shfl_xor(v, 1, 32);
        v += __shfl_xor(v, 2, 32);
        v += __shfl_xor(v, 4, 32);
        v += __shfl_xor(v, 8, 32);
        spart[i] = v;
    }
    if (lan16 == 0) {
        int r0 = sbase + wv * 16 + half * 8;
#pragma unroll
        for (int i = 0; i < 8; ++i)
            scores[(size_t)b * BA_S + r0 + i] = spart[i] + Vb0;
    }
}

// ---------------------------------------------------------------------------
// K3a: per-batch softmax stats (max, sum of exp)
// ---------------------------------------------------------------------------
__global__ void k_stats(const float* __restrict__ scores, float* __restrict__ stats) {
    __shared__ float red[256];
    int b = blockIdx.x, tid = threadIdx.x;
    const float* sp = scores + (size_t)b * BA_S;
    float m = -INFINITY;
    for (int i = tid; i < BA_S; i += 256) m = fmaxf(m, sp[i]);
    red[tid] = m; __syncthreads();
    for (int o = 128; o > 0; o >>= 1) {
        if (tid < o) red[tid] = fmaxf(red[tid], red[tid + o]);
        __syncthreads();
    }
    m = red[0]; __syncthreads();
    float s = 0.f;
    for (int i = tid; i < BA_S; i += 256) s += __expf(sp[i] - m);
    red[tid] = s; __syncthreads();
    for (int o = 128; o > 0; o >>= 1) {
        if (tid < o) red[tid] += red[tid + o];
        __syncthreads();
    }
    if (tid == 0) { stats[2 * b] = m; stats[2 * b + 1] = red[0]; }
}

// ---------------------------------------------------------------------------
// K3b: weights -> d_out, partial context over 512-row S chunk.
// Grid: 32*8 blocks x 256 threads. Coalesced streaming of encoder_outputs.
// ---------------------------------------------------------------------------
__global__ void k_weights_ctx(const float* __restrict__ scores,
                              const float* __restrict__ stats,
                              const float* __restrict__ enc,
                              float* __restrict__ attn_out,
                              float* __restrict__ ctx_part) {
    __shared__ float wlds[512];
    int b = blockIdx.x >> 3, ch = blockIdx.x & 7, tid = threadIdx.x;
    int s0 = ch * 512;
    float m = stats[2 * b];
    float inv = 1.f / stats[2 * b + 1];
    for (int i = tid; i < 512; i += 256) {
        float w = __expf(scores[(size_t)b * BA_S + s0 + i] - m) * inv;
        wlds[i] = w;
        attn_out[(size_t)b * BA_S + s0 + i] = w;
    }
    __syncthreads();
    float acc = 0.f;
    const float* ep = enc + ((size_t)b * BA_S + s0) * BA_H + tid;
#pragma unroll 4
    for (int s = 0; s < 512; ++s)
        acc += wlds[s] * ep[(size_t)s * BA_H];
    ctx_part[((size_t)b * 8 + ch) * BA_H + tid] = acc;
}

// ---------------------------------------------------------------------------
// K4: sum partial contexts -> d_out context region
// ---------------------------------------------------------------------------
__global__ void k_ctx_final(const float* __restrict__ ctx_part, float* __restrict__ ctx_out) {
    int b = blockIdx.x, tid = threadIdx.x;
    float s = 0.f;
#pragma unroll
    for (int ch = 0; ch < 8; ++ch)
        s += ctx_part[((size_t)b * 8 + ch) * BA_H + tid];
    ctx_out[(size_t)b * BA_H + tid] = s;
}

// ---------------------------------------------------------------------------
extern "C" void kernel_launch(void* const* d_in, const int* in_sizes, int n_in,
                              void* d_out, int out_size, void* d_ws, size_t ws_size,
                              hipStream_t stream) {
    const float* hidden = (const float*)d_in[0];   // (B,H)
    const float* enc    = (const float*)d_in[1];   // (B,S,H)
    const float* W1_w   = (const float*)d_in[2];   // (H,H)
    const float* W1_b   = (const float*)d_in[3];   // (H,)
    const float* W2_w   = (const float*)d_in[4];   // (H,H)
    const float* W2_b   = (const float*)d_in[5];   // (H,)
    const float* V_w    = (const float*)d_in[6];   // (H,1)
    const float* V_b    = (const float*)d_in[7];   // (1,)

    float* attn_out = (float*)d_out;                       // B*S
    float* ctx_out  = (float*)d_out + (size_t)BA_B * BA_S; // B*H

    char* ws = (char*)d_ws;
    float*          qb2      = (float*)(ws + 0);           //  32768 B
    unsigned short* pw2      = (unsigned short*)(ws + 32768);  // 131072 B
    float*          scores   = (float*)(ws + 163840);      // 524288 B
    float*          stats    = (float*)(ws + 688128);      //    256 B
    float*          ctx_part = (float*)(ws + 688384);      // 262144 B

    k_pack_w2   <<<256,  256, 0, stream>>>(W2_w, pw2);
    k_qb2       <<<BA_B, 256, 0, stream>>>(hidden, W1_w, W1_b, W2_b, qb2);
    k_scores    <<<BA_B * (BA_S / 128), 256, 0, stream>>>(enc, pw2, qb2, V_w, V_b, scores);
    k_stats     <<<BA_B, 256, 0, stream>>>(scores, stats);
    k_weights_ctx<<<BA_B * 8, 256, 0, stream>>>(scores, stats, enc, attn_out, ctx_part);
    k_ctx_final <<<BA_B, 256, 0, stream>>>(ctx_part, ctx_out);
}
